// HessianDiagonal2_67070209295171
// MI455X (gfx1250) — matrix operand order repaired and checked by reading, not compiled, not run on hardware
//
#include <hip/hip_runtime.h>
#include <hip/hip_bf16.h>

// ---------------------------------------------------------------------------
// MI455X (gfx1250) implementation.
//  - 3 dominant GEMMs on v_wmma_f32_16x16x32_f16 (f16 in, f32 acc)
//  - weights pre-converted+transposed to f16 in d_ws (fragment-friendly [n][K])
//  - float4-vectorized global->LDS staging (kernel is HBM-bound: ~420MB @ 23.3TB/s)
//  - SiLU fused into GEMM3 store epilogue; head dot-product wave-reduced
//  - fast v_sqrt_f32 / v_rcp_f32 for norm + SiLU (1-ulp, feeds f16 anyway)
// ---------------------------------------------------------------------------

typedef __attribute__((ext_vector_type(16))) _Float16 v16h;
typedef __attribute__((ext_vector_type(8)))  _Float16 v8h;
typedef __attribute__((ext_vector_type(4)))  _Float16 v4h;
typedef __attribute__((ext_vector_type(8)))  float    v8f;

#define N_ATOMS_C 200000
#define APB 32                 // atoms per block
#define ROWS (APB*3)           // 96 rows for GEMM1

// f16 weight workspace offsets (in halves)
#define WS_WMIX1 0                       // [128][128]  (WT[o][i] = Wmix1[i][o])
#define WS_W11   (128*128)               // [64][192]
#define WS_W21   (WS_W11 + 64*192)       // [128][64]
#define WS_TOTAL (WS_W21 + 128*64)       // 36864 halves = 73728 bytes

__device__ __forceinline__ float fast_sqrt(float x) {
    return __builtin_amdgcn_sqrtf(x);            // v_sqrt_f32
}
__device__ __forceinline__ float silu_f(float x) {
    // x * rcp(1 + exp(-x)) : v_exp_f32 + v_rcp_f32 (TRANS), no precise-div expansion
    return x * __builtin_amdgcn_rcpf(1.0f + __expf(-x));
}

union Frag16 { v16h v; v8h h[2]; };

// A-matrix 16x32 f16 fragment (ISA 7.12.2): lane m = lane&15;
// per lane: halves [kb..kb+7] and [kb+16..kb+23], kb = (lane>>4)*8 (+kstep*32)
__device__ __forceinline__ v16h load_fragA(const _Float16* row, int kb) {
    Frag16 f;
    f.h[0] = *(const v8h*)(row + kb);
    f.h[1] = *(const v8h*)(row + kb + 16);
    return f.v;
}

// B-matrix 32x16 f16 fragment: lane n = lane&15;
// per lane: halves [kb..kb+7] and [kb+16..kb+23], kb = (lane>>4)*8 (+kstep*32)
__device__ __forceinline__ v16h load_fragB(const _Float16* row, int kb) {
    Frag16 f;
    f.h[0] = *(const v8h*)(row + kb);
    f.h[1] = *(const v8h*)(row + kb + 16);
    return f.v;
}

// ---------------------------------------------------------------------------
// Prep: convert Wmix1/W1_1/W2_1 (fp32, [K][N] row-major) to f16 [N][K] in ws
// ---------------------------------------------------------------------------
__global__ void convert_weights_kernel(const float* __restrict__ Wmix1,
                                       const float* __restrict__ W11,
                                       const float* __restrict__ W21,
                                       _Float16* __restrict__ ws) {
    int t = blockIdx.x * blockDim.x + threadIdx.x;
    if (t < 128*128) {
        int o = t >> 7, i = t & 127;                     // K=128, N=128
        ws[WS_WMIX1 + o*128 + i] = (_Float16)Wmix1[i*128 + o];
    } else if (t < 128*128 + 64*192) {
        int u = t - 128*128;
        int o = u / 192, i = u % 192;                    // K=192, N=64
        ws[WS_W11 + o*192 + i] = (_Float16)W11[i*64 + o];
    } else if (t < WS_TOTAL) {
        int u = t - (128*128 + 64*192);
        int o = u >> 6, i = u & 63;                      // K=64, N=128
        ws[WS_W21 + o*64 + i] = (_Float16)W21[i*128 + o];
    }
}

// ---------------------------------------------------------------------------
// Main fused kernel: 32 atoms per 256-thread block (8 wave32s)
// ---------------------------------------------------------------------------
__global__ __launch_bounds__(256) void gated_blocks_kernel(
    const float* __restrict__ pos,   const float* __restrict__ srep,
    const float* __restrict__ vrep,  const _Float16* __restrict__ ws,
    const float* __restrict__ b11,   const float* __restrict__ b21,
    const float* __restrict__ Wmix2, const float* __restrict__ W12,
    const float* __restrict__ b12,   const float* __restrict__ W22,
    const float* __restrict__ b22,   const float* __restrict__ Wd1,
    const float* __restrict__ bd1,   const float* __restrict__ Wd2,
    const float* __restrict__ bd2,   float* __restrict__ out) {

    __shared__ __align__(16) _Float16 A1[ROWS*128];   // v tile, f16      (24 KB)
    __shared__ __align__(16) float    VM[ROWS*128];   // vmix = v@Wmix1   (48 KB)
    __shared__ __align__(16) _Float16 A2[APB*192];    // [s | vVn], f16   (12 KB)
    __shared__ __align__(16) _Float16 X1[APB*64];     // silu(h1), f16    ( 4 KB)
    __shared__ __align__(16) float    X2[APB*128];    // [silu(s_out)|gate] (16 KB)
    __shared__ float M2[APB*3*2];                     // block-2 mixes
    __shared__ float WM2[128];                        // Wmix2 (64x2)
    __shared__ float W12l[65];                        // W1_2
    __shared__ float SSUM[APB];                       // head dot partials

    const int t      = threadIdx.x;
    const int lane   = t & 31;
    const int wave   = t >> 5;
    const int lanelo = lane & 15;
    const int lanehi = lane >> 4;
    const int a0     = blockIdx.x * APB;
    const bool full  = (a0 + APB) <= N_ATOMS_C;

    // ---- stage in: v -> A1 f16 [96][128], s -> A2[:, 0:128] f16 ----
    if (full) {
        // v[a0 .. a0+31] is one contiguous run of 96*128 floats
        const float4* v4 = (const float4*)(vrep + (size_t)a0 * 384);
        for (int i = t; i < ROWS*32; i += 256) {        // 3072 float4 loads
            float4 x = v4[i];
            *(v4h*)&A1[i*4] = (v4h){(_Float16)x.x, (_Float16)x.y,
                                    (_Float16)x.z, (_Float16)x.w};
        }
        const float4* s4 = (const float4*)(srep + (size_t)a0 * 128);
        for (int i = t; i < APB*32; i += 256) {         // 1024 float4 loads
            float4 x = s4[i];
            int a = i >> 5, k4 = i & 31;
            *(v4h*)&A2[a*192 + k4*4] = (v4h){(_Float16)x.x, (_Float16)x.y,
                                             (_Float16)x.z, (_Float16)x.w};
        }
    } else {
        for (int i = t; i < ROWS*128; i += 256) {
            int row = i >> 7, k = i & 127;
            int a = row / 3;
            float val = (a0 + a < N_ATOMS_C) ? vrep[(size_t)(a0 + a)*384 + (row % 3)*128 + k] : 0.0f;
            A1[i] = (_Float16)val;
        }
        for (int i = t; i < APB*128; i += 256) {
            int a = i >> 7, k = i & 127;
            float val = (a0 + a < N_ATOMS_C) ? srep[(size_t)(a0 + a)*128 + k] : 0.0f;
            A2[a*192 + k] = (_Float16)val;
        }
    }
    if (t < 128) WM2[t] = Wmix2[t];
    else if (t < 128 + 65) W12l[t - 128] = W12[t - 128];
    __syncthreads();

    // ---- GEMM1: (96x128) @ (128x128) -> VM, 6 Mtiles x 8 Ntiles, K=128 ----
    for (int tt = wave; tt < 48; tt += 8) {
        int mt = tt >> 3, nt = tt & 7;
        v8f acc = {};
        const _Float16* arow = &A1[(mt*16 + lanelo)*128];
        const _Float16* brow = &ws[WS_WMIX1 + (nt*16 + lanelo)*128];
        #pragma unroll
        for (int ks = 0; ks < 4; ++ks) {
            v16h af = load_fragA(arow, lanehi*8  + ks*32);
            v16h bf = load_fragB(brow, lanehi*8  + ks*32);
            acc = __builtin_amdgcn_wmma_f32_16x16x32_f16(false, af, false, bf,
                                                         (short)0, acc, false, false);
        }
        #pragma unroll
        for (int j = 0; j < 8; ++j) {
            int m = mt*16 + j + lanehi*8;
            VM[m*128 + nt*16 + lanelo] = acc[j];
        }
    }
    __syncthreads();

    // ---- vVn = ||vmix[:, :, :64]|| over c -> A2[:, 128:192] ----
    for (int i = t; i < APB*64; i += 256) {
        int a = i >> 6, o = i & 63;
        float x0 = VM[(3*a + 0)*128 + o];
        float x1 = VM[(3*a + 1)*128 + o];
        float x2 = VM[(3*a + 2)*128 + o];
        A2[a*192 + 128 + o] = (_Float16)fast_sqrt(x0*x0 + x1*x1 + x2*x2);
    }
    __syncthreads();

    // ---- GEMM2: (32x192) @ (192x64) -> X1 = f16(silu(. + b1_1)) ----
    {   // 2 Mtiles x 4 Ntiles = 8 tiles, one per wave, K=192
        int mt = wave >> 2, nt = wave & 3;
        v8f acc = {};
        const _Float16* arow = &A2[(mt*16 + lanelo)*192];
        const _Float16* brow = &ws[WS_W11 + (nt*16 + lanelo)*192];
        #pragma unroll
        for (int ks = 0; ks < 6; ++ks) {
            v16h af = load_fragA(arow, lanehi*8  + ks*32);
            v16h bf = load_fragB(brow, lanehi*8  + ks*32);
            acc = __builtin_amdgcn_wmma_f32_16x16x32_f16(false, af, false, bf,
                                                         (short)0, acc, false, false);
        }
        int n = nt*16 + lanelo;
        float bias = b11[n];
        #pragma unroll
        for (int j = 0; j < 8; ++j) {
            int m = mt*16 + j + lanehi*8;
            X1[m*64 + n] = (_Float16)silu_f(acc[j] + bias);
        }
    }
    __syncthreads();

    // ---- GEMM3: (32x64) @ (64x128) -> X2; cols<64 get SiLU fused (s_out) ----
    for (int tt = wave; tt < 16; tt += 8) {   // 2 Mtiles x 8 Ntiles, K=64
        int mt = tt >> 3, nt = tt & 7;
        v8f acc = {};
        const _Float16* arow = &X1[(mt*16 + lanelo)*64];
        const _Float16* brow = &ws[WS_W21 + (nt*16 + lanelo)*64];
        #pragma unroll
        for (int ks = 0; ks < 2; ++ks) {
            v16h af = load_fragA(arow, lanehi*8  + ks*32);
            v16h bf = load_fragB(brow, lanehi*8  + ks*32);
            acc = __builtin_amdgcn_wmma_f32_16x16x32_f16(false, af, false, bf,
                                                         (short)0, acc, false, false);
        }
        int n = nt*16 + lanelo;
        float bias = b21[n];
        bool do_silu = (nt < 4);              // s_out half gets SiLU here
        #pragma unroll
        for (int j = 0; j < 8; ++j) {
            int m = mt*16 + j + lanehi*8;
            float val = acc[j] + bias;
            if (do_silu) val = silu_f(val);
            X2[m*128 + n] = val;
        }
    }
    __syncthreads();

    // ---- head dot: SSUM[a] = sum_o silu(s_out)[a,o] * W1_2[o], 8 lanes/atom ----
    {
        int a = t >> 3, oo = t & 7;           // 256 threads = 32 atoms x 8 lanes
        float p = 0.0f;
        #pragma unroll
        for (int j = 0; j < 8; ++j) {
            int o = oo + 8*j;
            p += X2[a*128 + o] * W12l[o];     // X2 cols<64 already silu'd
        }
        p += __shfl_xor(p, 1, 32);
        p += __shfl_xor(p, 2, 32);
        p += __shfl_xor(p, 4, 32);
        if (oo == 0) SSUM[a] = p;
    }

    // ---- block 2 vector mix: per (atom, component), K=64, Nout=2 ----
    if (t < APB*3) {
        int a = t / 3, c = t % 3;
        float m0 = 0.0f, m1 = 0.0f;
        for (int o = 0; o < 64; ++o) {
            float g  = X2[a*128 + 64 + o];          // gate (raw)
            float vw = VM[(3*a + c)*128 + 64 + o];  // vW
            float vo = g * vw;                      // v_out of block 1
            m0 += vo * WM2[o*2 + 0];
            m1 += vo * WM2[o*2 + 1];
        }
        M2[t*2 + 0] = m0;
        M2[t*2 + 1] = m1;
    }
    __syncthreads();

    // ---- per-atom head: block-2 scalar path + 7->10->3 MLP ----
    if (t < APB && (a0 + t) < N_ATOMS_C) {
        int a = t;
        float v0 = M2[(a*3 + 0)*2], v1 = M2[(a*3 + 1)*2], v2 = M2[(a*3 + 2)*2];
        float vvn2 = fast_sqrt(v0*v0 + v1*v1 + v2*v2);
        float h = b12[0] + SSUM[a] + vvn2 * W12l[64];
        float x = silu_f(h);
        float sout2 = x*W22[0] + b22[0];            // l0 (no silu)
        float gate2 = x*W22[1] + b22[1];

        float feat[7];
        feat[0] = gate2 * M2[(a*3 + 0)*2 + 1];      // l1
        feat[1] = gate2 * M2[(a*3 + 1)*2 + 1];
        feat[2] = gate2 * M2[(a*3 + 2)*2 + 1];
        feat[3] = pos[(size_t)(a0 + a)*3 + 0];
        feat[4] = pos[(size_t)(a0 + a)*3 + 1];
        feat[5] = pos[(size_t)(a0 + a)*3 + 2];
        feat[6] = sout2;

        float hid[10];
        #pragma unroll
        for (int o = 0; o < 10; ++o) {
            float acc2 = bd1[o];
            #pragma unroll
            for (int i = 0; i < 7; ++i) acc2 += feat[i] * Wd1[i*10 + o];
            hid[o] = silu_f(acc2);
        }
        #pragma unroll
        for (int j = 0; j < 3; ++j) {
            float acc2 = bd2[j];
            #pragma unroll
            for (int o = 0; o < 10; ++o) acc2 += hid[o] * Wd2[o*3 + j];
            out[(size_t)(a0 + a)*3 + j] = acc2;
        }
    }
}

// ---------------------------------------------------------------------------
extern "C" void kernel_launch(void* const* d_in, const int* in_sizes, int n_in,
                              void* d_out, int out_size, void* d_ws, size_t ws_size,
                              hipStream_t stream) {
    const float* pos   = (const float*)d_in[0];
    const float* srep  = (const float*)d_in[1];
    const float* vrep  = (const float*)d_in[2];
    // d_in[3] = idx_m (unused by reference output)
    const float* Wmix1 = (const float*)d_in[4];
    const float* W11   = (const float*)d_in[5];
    const float* b11   = (const float*)d_in[6];
    const float* W21   = (const float*)d_in[7];
    const float* b21   = (const float*)d_in[8];
    const float* Wmix2 = (const float*)d_in[9];
    const float* W12   = (const float*)d_in[10];
    const float* b12   = (const float*)d_in[11];
    const float* W22   = (const float*)d_in[12];
    const float* b22   = (const float*)d_in[13];
    const float* Wd1   = (const float*)d_in[14];
    const float* bd1   = (const float*)d_in[15];
    const float* Wd2   = (const float*)d_in[16];
    const float* bd2   = (const float*)d_in[17];
    _Float16* ws  = (_Float16*)d_ws;
    float*    out = (float*)d_out;

    convert_weights_kernel<<<(WS_TOTAL + 255)/256, 256, 0, stream>>>(Wmix1, W11, W21, ws);

    int nblocks = (N_ATOMS_C + APB - 1) / APB;   // 6250
    gated_blocks_kernel<<<nblocks, 256, 0, stream>>>(
        pos, srep, vrep, ws, b11, b21, Wmix2, W12, b12, W22, b22,
        Wd1, bd1, Wd2, bd2, out);
}
